// Mutihead_Attention_19267223290336
// MI455X (gfx1250) — matrix operand order repaired and checked by reading, not compiled, not run on hardware
//
#include <hip/hip_runtime.h>

typedef __attribute__((ext_vector_type(16))) __bf16 v16bf;
typedef __attribute__((ext_vector_type(8)))  __bf16 v8bf;
typedef __attribute__((ext_vector_type(8)))  float  v8f;

constexpr int SEQ    = 2048;
constexpr int DMODEL = 512;
constexpr int DHEAD  = 64;
constexpr int NSLAB  = 16;                    // H * B
constexpr float NORM = 0.04419417382415922f;  // 1/sqrt(512)

constexpr int AST = 40;  // padded LDS stride (bf16) for 32-wide GEMM tiles

static __device__ __forceinline__ __bf16 f2bf(float f) {
  unsigned u = __builtin_bit_cast(unsigned, f);
  unsigned r = u + 0x7FFFu + ((u >> 16) & 1u);  // RNE
  unsigned short h = (unsigned short)(r >> 16);
  return __builtin_bit_cast(__bf16, h);
}

// monotone fp32 <-> uint key maps (key order == float order)
static __device__ __forceinline__ unsigned f2key(unsigned u) {
  return u ^ ((u >> 31) ? 0xFFFFFFFFu : 0x80000000u);
}
static __device__ __forceinline__ float key2f(unsigned u) {
  unsigned b = (u & 0x80000000u) ? (u ^ 0x80000000u) : ~u;
  return __builtin_bit_cast(float, b);
}

static __device__ __forceinline__ v16bf mk16(v8bf lo, v8bf hi) {
  return __builtin_shufflevector(lo, hi, 0, 1, 2, 3, 4, 5, 6, 7,
                                 8, 9, 10, 11, 12, 13, 14, 15);
}

// ---------------------------------------------------------------------------
// C[M,N] = A[M,K] @ W[K,N] + bias, ping-pong double-buffered LDS.
// OMODE: 0 = bf16 flat out, 1 = fp32 flat out, 2 = bf16 transposed-slab out
//        (Vt[hb][d][s], for barrier-free P@V B-fragments).
// Block: 128x64 tile, 8 waves (4x2), each wave 32x32 = 2x2 WMMA accumulators.
// ---------------------------------------------------------------------------
template <bool AF32, int OMODE>
__global__ __launch_bounds__(256) void gemm_bf16_wmma(
    const void* __restrict__ Ap, const float* __restrict__ W,
    const float* __restrict__ bias, void* __restrict__ Outp,
    int M, int N, int K) {
  __shared__ __bf16 As[2][128 * AST];
  __shared__ __bf16 Bs[2][64 * AST];

  const int tid  = threadIdx.x;
  const int lane = tid & 31, wave = tid >> 5;
  const int lm = lane & 15, hi = lane >> 4;
  const int m0 = blockIdx.y * 128, n0 = blockIdx.x * 64;
  const int wm = (wave & 3) * 32, wn = (wave >> 2) * 32;

  auto stage = [&](int buf, int k0) {
    // A tile 128x32 -> bf16 LDS
    for (int i = tid * 4; i < 128 * 32; i += 256 * 4) {
      int r = i >> 5, c = i & 31;
      if constexpr (AF32) {
        const float* ap = (const float*)Ap + (size_t)(m0 + r) * K + k0 + c;
#pragma unroll
        for (int j = 0; j < 4; ++j) As[buf][r * AST + c + j] = f2bf(ap[j]);
      } else {
        const __bf16* ap = (const __bf16*)Ap + (size_t)(m0 + r) * K + k0 + c;
#pragma unroll
        for (int j = 0; j < 4; ++j) As[buf][r * AST + c + j] = ap[j];
      }
    }
    // W tile 32x64 transposed -> Bs[n][k] (bf16)
    for (int i = tid; i < 32 * 64; i += 256) {
      int k = i >> 6, n = i & 63;
      Bs[buf][n * AST + k] = f2bf(W[(size_t)(k0 + k) * N + n0 + n]);
    }
  };

  v8f acc[2][2] = {};
  const int nk = K >> 5;

  stage(0, 0);
  __syncthreads();
  for (int it = 0; it < nk; ++it) {
    const int cur = it & 1;
    if (it + 1 < nk) stage(cur ^ 1, (it + 1) * 32);  // overlap with WMMA below

    v16bf a[2], b[2];
#pragma unroll
    for (int mt = 0; mt < 2; ++mt) {
      const __bf16* ar = &As[cur][(wm + mt * 16 + lm) * AST];
      a[mt] = mk16(*(const v8bf*)(ar + hi * 8), *(const v8bf*)(ar + 16 + hi * 8));
    }
#pragma unroll
    for (int nt = 0; nt < 2; ++nt) {
      const __bf16* br = &Bs[cur][(wn + nt * 16 + lm) * AST];
      b[nt] = mk16(*(const v8bf*)(br + hi * 8), *(const v8bf*)(br + 16 + hi * 8));
    }
#pragma unroll
    for (int mt = 0; mt < 2; ++mt)
#pragma unroll
      for (int nt = 0; nt < 2; ++nt)
        acc[mt][nt] = __builtin_amdgcn_wmma_f32_16x16x32_bf16(
            false, a[mt], false, b[nt], (short)0, acc[mt][nt], false, false);
    __syncthreads();
  }

#pragma unroll
  for (int mt = 0; mt < 2; ++mt)
#pragma unroll
    for (int nt = 0; nt < 2; ++nt) {
      int n = n0 + wn + nt * 16 + lm;
      float bv = bias[n];
#pragma unroll
      for (int r = 0; r < 8; ++r) {
        int m = m0 + wm + mt * 16 + r + 8 * hi;
        float v = acc[mt][nt][r] + bv;
        if constexpr (OMODE == 1) {
          ((float*)Outp)[(size_t)m * N + n] = v;
        } else if constexpr (OMODE == 0) {
          ((__bf16*)Outp)[(size_t)m * N + n] = f2bf(v);
        } else {
          // transposed slab layout: flat idx -> (hb, s, d) -> Vt[hb][d][s]
          size_t idx = (size_t)m * N + n;
          size_t hb  = idx >> 17;  // / (SEQ*DHEAD)
          int    rem = (int)(idx & 131071);
          int    s   = rem >> 6, d = rem & 63;
          ((__bf16*)Outp)[(hb << 17) + (size_t)d * SEQ + s] = f2bf(v);
        }
      }
    }
}

// ---------------------------------------------------------------------------
// Fused attention: per (slab, 16-query tile):
//   scores (WMMA, LDS strip) -> register-resident median+softmax -> P@V (WMMA,
//   barrier-free, B-fragments streamed from transposed global Vt).
// ---------------------------------------------------------------------------
__global__ __launch_bounds__(256) void attn_median_softmax(
    const unsigned short* __restrict__ Qb_, const unsigned short* __restrict__ Kb_,
    const unsigned short* __restrict__ Vt_, unsigned short* __restrict__ Ob_) {
  extern __shared__ char smem[];
  float*  sc  = (float*)smem;                    // 16 x 2048 f32 (128KB)
  __bf16* pbf = (__bf16*)(smem + 16 * SEQ * 4);  // 16 x 2048 bf16 (64KB)
  float*  po  = sc;  // reuse score region for cross-wave partial-O exchange

  const int tid = threadIdx.x, lane = tid & 31, wave = tid >> 5;
  const int lm = lane & 15, hi = lane >> 4;
  const int hb = blockIdx.y;
  const int q0 = blockIdx.x * 16;

  const __bf16* Qs = (const __bf16*)Qb_ + (size_t)hb * SEQ * DHEAD;
  const __bf16* Ks = (const __bf16*)Kb_ + (size_t)hb * SEQ * DHEAD;
  __bf16*       Os = (__bf16*)Ob_ + (size_t)hb * SEQ * DHEAD;

  // ---- Phase 1: scores = (Q @ K^T) * NORM; wave w owns key cols [w*256, +256)
  v16bf aq[2];
  {
    const __bf16* qr = Qs + (size_t)(q0 + lm) * DHEAD;
#pragma unroll
    for (int ks = 0; ks < 2; ++ks) {
      const __bf16* p = qr + ks * 32;
      aq[ks] = mk16(*(const v8bf*)(p + hi * 8), *(const v8bf*)(p + 16 + hi * 8));
    }
  }
  for (int nt = 0; nt < 16; ++nt) {
    int n0c = wave * 256 + nt * 16;
    const __bf16* kr = Ks + (size_t)(n0c + lm) * DHEAD;
    if (nt + 1 < 16)  // warm next K tile (global_prefetch_b8)
      __builtin_prefetch(kr + 16 * DHEAD, 0, 3);
    v8f acc = {};
#pragma unroll
    for (int ks = 0; ks < 2; ++ks) {
      const __bf16* p = kr + ks * 32;
      v16bf bk = mk16(*(const v8bf*)(p + hi * 8), *(const v8bf*)(p + 16 + hi * 8));
      acc = __builtin_amdgcn_wmma_f32_16x16x32_bf16(false, aq[ks], false, bk,
                                                    (short)0, acc, false, false);
    }
#pragma unroll
    for (int r = 0; r < 8; ++r)
      sc[(r + 8 * hi) * SEQ + n0c + lm] = acc[r] * NORM;
  }
  __syncthreads();

  // ---- Phase 2+3 fused (register resident): lower median (rank 1023/2048)
  //      via 32-step radix bisection on monotone keys, then masked softmax.
  //      One wave per row; each lane caches 64 elements (loaded once as b64).
  for (int row = wave; row < 16; row += 8) {
    const float* rp = sc + row * SEQ;
    unsigned key[64];
#pragma unroll
    for (int j = 0; j < 32; ++j) {
      float2 f2 = *(const float2*)(rp + j * 64 + lane * 2);
      key[2 * j]     = f2key(__builtin_bit_cast(unsigned, f2.x));
      key[2 * j + 1] = f2key(__builtin_bit_cast(unsigned, f2.y));
    }
    unsigned prefix = 0;
    int k = 1023;
    for (int bit = 31; bit >= 0; --bit) {
      int c = 0;
#pragma unroll
      for (int j = 0; j < 64; ++j) c += (((key[j] ^ prefix) >> bit) == 0u);
#pragma unroll
      for (int off = 16; off > 0; off >>= 1) c += __shfl_xor(c, off, 32);
      if (k >= c) { k -= c; prefix |= (1u << bit); }
    }
    // survivors: key > prefix (i.e. score > median). Row max in key space.
    unsigned mk = 0u;
#pragma unroll
    for (int j = 0; j < 64; ++j)
      if (key[j] > prefix && key[j] > mk) mk = key[j];
#pragma unroll
    for (int off = 16; off > 0; off >>= 1) {
      unsigned o = (unsigned)__shfl_xor((int)mk, off, 32);
      if (o > mk) mk = o;
    }
    float mx = key2f(mk);
    float ev[64];
    float sum = 0.f;
#pragma unroll
    for (int j = 0; j < 64; ++j) {
      float e = (key[j] > prefix) ? __expf(key2f(key[j]) - mx) : 0.f;
      ev[j] = e;
      sum += e;
    }
#pragma unroll
    for (int off = 16; off > 0; off >>= 1) sum += __shfl_xor(sum, off, 32);
    float rinv = 1.0f / sum;
    __bf16* prow = pbf + row * SEQ;
#pragma unroll
    for (int j = 0; j < 32; ++j) {  // packed 2 x bf16 per b32 store
      unsigned lo = (unsigned)__builtin_bit_cast(unsigned short, f2bf(ev[2 * j] * rinv));
      unsigned hi2 = (unsigned)__builtin_bit_cast(unsigned short, f2bf(ev[2 * j + 1] * rinv));
      *(unsigned*)(prow + j * 64 + lane * 2) = lo | (hi2 << 16);
    }
  }
  __syncthreads();

  // ---- Phase 4: O = P @ V, barrier-free. wave -> (n-tile = wave&3, k-half).
  //      B-fragments are contiguous rows of global Vt[hb][d][s].
  //      Two independent accumulators break the serial D->C WMMA chain.
  const int nt = wave & 3, kh = wave >> 2;
  const __bf16* vrow = (const __bf16*)Vt_ + ((size_t)hb << 17) +
                       (size_t)(nt * 16 + lm) * SEQ;
  v8f oa0 = {}, oa1 = {};
  for (int k0 = kh * 1024; k0 < kh * 1024 + 1024; k0 += 64) {
    {
      const __bf16* pr = pbf + lm * SEQ + k0;
      v16bf ap = mk16(*(const v8bf*)(pr + hi * 8), *(const v8bf*)(pr + 16 + hi * 8));
      const __bf16* vr = vrow + k0;
      __builtin_prefetch(vr + 128, 0, 3);
      v16bf bv = mk16(*(const v8bf*)(vr + hi * 8), *(const v8bf*)(vr + 16 + hi * 8));
      oa0 = __builtin_amdgcn_wmma_f32_16x16x32_bf16(false, ap, false, bv,
                                                    (short)0, oa0, false, false);
    }
    {
      const __bf16* pr = pbf + lm * SEQ + k0 + 32;
      v16bf ap = mk16(*(const v8bf*)(pr + hi * 8), *(const v8bf*)(pr + 16 + hi * 8));
      const __bf16* vr = vrow + k0 + 32;
      v16bf bv = mk16(*(const v8bf*)(vr + hi * 8), *(const v8bf*)(vr + 16 + hi * 8));
      oa1 = __builtin_amdgcn_wmma_f32_16x16x32_bf16(false, ap, false, bv,
                                                    (short)0, oa1, false, false);
    }
  }
  v8f oacc = oa0 + oa1;
  if (kh == 1) {
#pragma unroll
    for (int r = 0; r < 8; ++r)
      po[(r + 8 * hi) * 64 + nt * 16 + lm] = oacc[r];
  }
  __syncthreads();
  if (kh == 0) {
#pragma unroll
    for (int r = 0; r < 8; ++r) {
      int m = r + 8 * hi;
      float v = oacc[r] + po[m * 64 + nt * 16 + lm];
      Os[(size_t)(q0 + m) * DHEAD + nt * 16 + lm] = f2bf(v);
    }
  }
}

// ---------------------------------------------------------------------------
extern "C" void kernel_launch(void* const* d_in, const int* in_sizes, int n_in,
                              void* d_out, int out_size, void* d_ws, size_t ws_size,
                              hipStream_t stream) {
  (void)in_sizes; (void)n_in; (void)out_size; (void)ws_size;
  const float* x  = (const float*)d_in[0];
  const float* y  = (const float*)d_in[1];
  const float* Wq = (const float*)d_in[2];
  const float* bq = (const float*)d_in[3];
  const float* Wk = (const float*)d_in[4];
  const float* bk = (const float*)d_in[5];
  const float* Wv = (const float*)d_in[6];
  const float* bv = (const float*)d_in[7];
  const float* Wo = (const float*)d_in[8];
  const float* bo = (const float*)d_in[9];

  const int M = 2 * SEQ;                    // 4096 rows (B*S)
  const size_t slab = (size_t)M * DMODEL;   // 2M elements
  unsigned short* Qb = (unsigned short*)d_ws;
  unsigned short* Kb = Qb + slab;
  unsigned short* Vt = Kb + slab;           // transposed slab layout
  unsigned short* Ob = Vt + slab;

  dim3 gg(DMODEL / 64, M / 128);  // (8, 32)
  gemm_bf16_wmma<true, 0><<<gg, 256, 0, stream>>>(x, Wq, bq, Qb, M, DMODEL, DMODEL);
  gemm_bf16_wmma<true, 0><<<gg, 256, 0, stream>>>(y, Wk, bk, Kb, M, DMODEL, DMODEL);
  gemm_bf16_wmma<true, 2><<<gg, 256, 0, stream>>>(y, Wv, bv, Vt, M, DMODEL, DMODEL);

  size_t smem = (size_t)16 * SEQ * 4    // fp32 score strip
              + (size_t)16 * SEQ * 2;   // bf16 probabilities
  attn_median_softmax<<<dim3(SEQ / 16, NSLAB), 256, smem, stream>>>(Qb, Kb, Vt, Ob);

  gemm_bf16_wmma<false, 1><<<gg, 256, 0, stream>>>(Ob, Wo, bo, d_out, M, DMODEL, DMODEL);
}
